// MLP_40097814675505
// MI455X (gfx1250) — matrix operand order repaired and checked by reading, not compiled, not run on hardware
//
#include <hip/hip_runtime.h>
#include <math.h>

// ---------------------------------------------------------------------------
// BitLinear MLP for MI455X (gfx1250):  out = bitlin2(gelu(bitlin1(x,w1)), w2)
// M = 8192, D = 2048, F = 8192.  GEMMs run on v_wmma_f32_16x16x32_bf16 with
// fp32 accumulate; weight sign matrices are exactly +/-1 in bf16, beta applied
// in the fp32 epilogue.  Tile staging uses the Tensor Data Mover
// (tensor_load_to_lds + s_wait_tensorcnt) when the builtin is available,
// with a global_load_b128 -> ds_store_b128 fallback otherwise.
// ---------------------------------------------------------------------------

typedef __attribute__((ext_vector_type(16))) __bf16 v16bf;
typedef __attribute__((ext_vector_type(8)))  float  v8f;
typedef unsigned int v4u  __attribute__((ext_vector_type(4)));
typedef float        v4f  __attribute__((ext_vector_type(4)));
typedef unsigned int u32x4 __attribute__((ext_vector_type(4)));
typedef int          i32x4 __attribute__((ext_vector_type(4)));
typedef int          i32x8 __attribute__((ext_vector_type(8)));

#define M_DIM 8192
#define D_DIM 2048
#define F_DIM 8192

#if defined(__gfx1250__) && __has_builtin(__builtin_amdgcn_tensor_load_to_lds) && \
    __has_builtin(__builtin_amdgcn_s_wait_tensorcnt)
#define USE_TDM 1
#else
#define USE_TDM 0
#endif

// ---- bf16 helpers (manual RNE; buffers are unsigned short) -----------------
static __device__ __forceinline__ unsigned short f2bf(float f) {
  unsigned int u = __builtin_bit_cast(unsigned int, f);
  u += 0x7fffu + ((u >> 16) & 1u);
  return (unsigned short)(u >> 16);
}
static __device__ __forceinline__ float bf2f(unsigned short h) {
  return __builtin_bit_cast(float, (unsigned int)h << 16);
}
static __device__ __forceinline__ unsigned int pack2(float a, float b) {
  return (unsigned int)f2bf(a) | ((unsigned int)f2bf(b) << 16);
}

struct Frag32 { v4u lo, hi; };
static __device__ __forceinline__ v16bf frag_cast(v4u lo, v4u hi) {
  Frag32 f{lo, hi};
  return __builtin_bit_cast(v16bf, f);
}

// ---------------------------------------------------------------------------
// Deterministic two-pass weight reduction:  sum(w), sum(|w|)
// ---------------------------------------------------------------------------
__global__ __launch_bounds__(256) void reduce_w(const float* __restrict__ w,
                                                int n,
                                                float* __restrict__ partials) {
  __shared__ float s1[256], s2[256];
  float sum = 0.f, sab = 0.f;
  for (int i = blockIdx.x * 256 + threadIdx.x; i < n; i += gridDim.x * 256) {
    float v = w[i];
    sum += v;
    sab += fabsf(v);
  }
  s1[threadIdx.x] = sum;
  s2[threadIdx.x] = sab;
  __syncthreads();
  for (int s = 128; s > 0; s >>= 1) {
    if (threadIdx.x < s) {
      s1[threadIdx.x] += s1[threadIdx.x + s];
      s2[threadIdx.x] += s2[threadIdx.x + s];
    }
    __syncthreads();
  }
  if (threadIdx.x == 0) {
    partials[2 * blockIdx.x]     = s1[0];
    partials[2 * blockIdx.x + 1] = s2[0];
  }
}

__global__ __launch_bounds__(256) void finalize_w(const float* __restrict__ partials,
                                                  int nblk, float inv_n,
                                                  float* __restrict__ out2) {
  __shared__ float s1[256], s2[256];
  float sum = 0.f, sab = 0.f;
  for (int i = threadIdx.x; i < nblk; i += 256) {
    sum += partials[2 * i];
    sab += partials[2 * i + 1];
  }
  s1[threadIdx.x] = sum;
  s2[threadIdx.x] = sab;
  __syncthreads();
  for (int s = 128; s > 0; s >>= 1) {
    if (threadIdx.x < s) {
      s1[threadIdx.x] += s1[threadIdx.x + s];
      s2[threadIdx.x] += s2[threadIdx.x + s];
    }
    __syncthreads();
  }
  if (threadIdx.x == 0) {
    out2[0] = s1[0] * inv_n;  // mean(w)
    out2[1] = s2[0] * inv_n;  // beta = mean(|w|)
  }
}

// sign(w - mean) as exact bf16 +/-1  (0x3F80 / 0xBF80)
__global__ __launch_bounds__(256) void quant_w(const float* __restrict__ w,
                                               const float* __restrict__ stats,
                                               unsigned short* __restrict__ ws,
                                               int n) {
  const float mean = stats[0];
  int i = (blockIdx.x * 256 + threadIdx.x) * 4;
  if (i + 3 < n) {
    v4f v = *(const v4f*)(w + i);
    unsigned int lo = ((v.x - mean) > 0.f ? 0x3F80u : 0xBF80u) |
                      (((v.y - mean) > 0.f ? 0x3F80u : 0xBF80u) << 16);
    unsigned int hi = ((v.z - mean) > 0.f ? 0x3F80u : 0xBF80u) |
                      (((v.w - mean) > 0.f ? 0x3F80u : 0xBF80u) << 16);
    unsigned long long p = (unsigned long long)lo | ((unsigned long long)hi << 32);
    *(unsigned long long*)(ws + i) = p;
  }
}

// ---------------------------------------------------------------------------
// Activation: layer_norm + absmax fake-quant of x rows (D=2048) -> bf16
// ---------------------------------------------------------------------------
__global__ __launch_bounds__(256) void act_quant_x(const float* __restrict__ x,
                                                   unsigned short* __restrict__ xq) {
  const int row = blockIdx.x;
  const int tid = threadIdx.x;
  const float* xr = x + (size_t)row * D_DIM;
  __shared__ float s1[256], s2[256];

  v4f a = ((const v4f*)xr)[tid * 2];
  v4f b = ((const v4f*)xr)[tid * 2 + 1];
  float v[8] = {a.x, a.y, a.z, a.w, b.x, b.y, b.z, b.w};

  float sum = 0.f, sq = 0.f;
#pragma unroll
  for (int j = 0; j < 8; ++j) { sum += v[j]; sq += v[j] * v[j]; }
  s1[tid] = sum; s2[tid] = sq;
  __syncthreads();
  for (int s = 128; s > 0; s >>= 1) {
    if (tid < s) { s1[tid] += s1[tid + s]; s2[tid] += s2[tid + s]; }
    __syncthreads();
  }
  const float mean = s1[0] * (1.0f / D_DIM);
  const float var  = s2[0] * (1.0f / D_DIM) - mean * mean;
  const float inv  = rsqrtf(var + 1e-6f);
  __syncthreads();

  float mx = 0.f;
#pragma unroll
  for (int j = 0; j < 8; ++j) {
    v[j] = (v[j] - mean) * inv;
    mx = fmaxf(mx, fabsf(v[j]));
  }
  s1[tid] = mx;
  __syncthreads();
  for (int s = 128; s > 0; s >>= 1) {
    if (tid < s) s1[tid] = fmaxf(s1[tid], s1[tid + s]);
    __syncthreads();
  }
  const float gamma = s1[0];
  const float up =  128.0f - 1e-6f;
  const float lo = -128.0f + 1e-6f;
  const float sA = 128.0f / gamma, sB = gamma * (1.0f / 128.0f);

  unsigned int o[4];
#pragma unroll
  for (int q = 0; q < 4; ++q) {
    float e0 = fminf(fmaxf(v[2 * q] * sA, lo), up) * sB;
    float e1 = fminf(fmaxf(v[2 * q + 1] * sA, lo), up) * sB;
    o[q] = pack2(e0, e1);
  }
  v4u out; out.x = o[0]; out.y = o[1]; out.z = o[2]; out.w = o[3];
  ((v4u*)(xq + (size_t)row * D_DIM))[tid] = out;
}

// ---------------------------------------------------------------------------
// Activation: layer_norm + absmax of h rows (F=8192, bf16, in place)
// ---------------------------------------------------------------------------
__global__ __launch_bounds__(256) void act_quant_h(unsigned short* __restrict__ h) {
  const int row = blockIdx.x;
  const int tid = threadIdx.x;
  unsigned short* hr = h + (size_t)row * F_DIM;
  __shared__ float buf[F_DIM];
  __shared__ float s1[256], s2[256];

  float sum = 0.f, sq = 0.f;
  const v4u* g = (const v4u*)hr;
#pragma unroll
  for (int j = 0; j < 4; ++j) {
    v4u u = g[tid * 4 + j];
    unsigned int w[4] = {u.x, u.y, u.z, u.w};
#pragma unroll
    for (int q = 0; q < 4; ++q) {
      float f0 = bf2f((unsigned short)(w[q] & 0xffffu));
      float f1 = bf2f((unsigned short)(w[q] >> 16));
      buf[tid * 32 + j * 8 + q * 2]     = f0;
      buf[tid * 32 + j * 8 + q * 2 + 1] = f1;
      sum += f0 + f1;
      sq  += f0 * f0 + f1 * f1;
    }
  }
  s1[tid] = sum; s2[tid] = sq;
  __syncthreads();
  for (int s = 128; s > 0; s >>= 1) {
    if (tid < s) { s1[tid] += s1[tid + s]; s2[tid] += s2[tid + s]; }
    __syncthreads();
  }
  const float mean = s1[0] * (1.0f / F_DIM);
  const float var  = s2[0] * (1.0f / F_DIM) - mean * mean;
  const float inv  = rsqrtf(var + 1e-6f);
  __syncthreads();

  float mx = 0.f;
#pragma unroll
  for (int j = 0; j < 32; ++j) {
    float xn = (buf[tid * 32 + j] - mean) * inv;
    mx = fmaxf(mx, fabsf(xn));
  }
  s1[tid] = mx;
  __syncthreads();
  for (int s = 128; s > 0; s >>= 1) {
    if (tid < s) s1[tid] = fmaxf(s1[tid], s1[tid + s]);
    __syncthreads();
  }
  const float gamma = s1[0];
  const float up =  128.0f - 1e-6f;
  const float lo = -128.0f + 1e-6f;
  const float sA = 128.0f / gamma, sB = gamma * (1.0f / 128.0f);

#pragma unroll
  for (int j = 0; j < 4; ++j) {
    unsigned int w[4];
#pragma unroll
    for (int q = 0; q < 4; ++q) {
      float x0 = (buf[tid * 32 + j * 8 + q * 2]     - mean) * inv;
      float x1 = (buf[tid * 32 + j * 8 + q * 2 + 1] - mean) * inv;
      x0 = fminf(fmaxf(x0 * sA, lo), up) * sB;
      x1 = fminf(fmaxf(x1 * sA, lo), up) * sB;
      w[q] = pack2(x0, x1);
    }
    v4u out; out.x = w[0]; out.y = w[1]; out.z = w[2]; out.w = w[3];
    ((v4u*)hr)[tid * 4 + j] = out;
  }
}

// ---------------------------------------------------------------------------
// WMMA GEMM:  C[m][n] = beta * sum_k A[m][k] * B[n][k]   (A,B bf16; B = +/-1)
// Block tile 128x256x32, 256 threads = 8 waves (2 M x 4 N), wave tile 64x64.
// LDS rows padded to 80B (40 bf16) -> conflict-free ds_load_b128.
// Staging: TDM tensor_load_to_lds (hardware pad) or global->ds fallback.
// ---------------------------------------------------------------------------
#define BM 128
#define BN 256
#define BK 32
#define LDSROW 40  // bf16 elements per padded row (80 bytes, 16B aligned)

#if USE_TDM
// Build a D# (ISA 8.3-8.5) for a rows x 32 bf16 tile of a row-major [rows][K]
// tensor and issue TENSOR_LOAD_TO_LDS.  data_size=1 (2B elements).
// pad_enable: every 16 DWORDs (64B = one 32-elem row) insert 4 DWORDs (16B)
// -> LDS row pitch = 80B = LDSROW elements.
// This toolchain's builtin takes 6 args:
//   (u32x4 group0, i32x8 group1, i32x4 group2, i32x4 group3, i32x8 extra, i32 cpol)
// The 5th operand corresponds to descriptor state beyond groups 2/3 (VADDR4
// is unused/NULL per the VIMAGE encoding) and is passed zero-filled.
static __device__ __forceinline__ void tdm_load_tile(const unsigned short* tile,
                                                     unsigned lds_addr,
                                                     int tile_rows, int K) {
  unsigned long long ga = (unsigned long long)tile;
  u32x4 g0;
  g0.x = 1u;                                   // count=1, user descriptor
  g0.y = lds_addr;                             // lds_addr[31:0]
  g0.z = (unsigned)ga;                         // global_addr[31:0]
  g0.w = (unsigned)((ga >> 32) & 0x01FFFFFFu)  // global_addr[56:32]
         | 0x80000000u;                        // type=2 ("image")
  i32x8 g1;
  g1[0] = (int)((1u << 16)      // data_size = 1 -> 2-byte elements
                | (1u << 20)    // pad_enable
                | (3u << 22)    // pad_interval: 2^(3+1)=16 DWORDs
                | (3u << 25));  // pad_amount: 3+1 = 4 DWORDs
  g1[1] = (int)(((unsigned)K & 0xFFFFu) << 16);          // tensor_dim0 lo
  g1[2] = (int)(((unsigned)K >> 16) | (0u << 16));       // tensor_dim0 hi | tensor_dim1 lo
  g1[3] = (int)((0x4000u) | (32u << 16));                // tensor_dim1 hi (big) | tile_dim0=32
  g1[4] = tile_rows;                                     // tile_dim1 (tile_dim2=0)
  g1[5] = K;                                             // tensor_dim0_stride lo
  g1[6] = 0;                                             // stride hi | dim1_stride lo
  g1[7] = 0;
  i32x4 g2; g2[0] = 1; g2[1] = 0; g2[2] = 0; g2[3] = 0;  // tensor_dim2=1, rest unused
  i32x4 g3; g3[0] = 0; g3[1] = 0; g3[2] = 0; g3[3] = 0;
  i32x8 g4;
  g4[0] = 0; g4[1] = 0; g4[2] = 0; g4[3] = 0;
  g4[4] = 0; g4[5] = 0; g4[6] = 0; g4[7] = 0;
  __builtin_amdgcn_tensor_load_to_lds(g0, g1, g2, g3, g4, 0);
}
static __device__ __forceinline__ unsigned lds_offset(const void* p) {
  // generic LDS pointers carry the LDS byte offset in their low 32 bits
  return (unsigned)(unsigned long long)p;
}
#endif

template <bool GELU, typename OutT>
__global__ __launch_bounds__(256) void gemm_bitlinear(
    const unsigned short* __restrict__ A,   // [M][K] bf16 activations
    const unsigned short* __restrict__ Bm,  // [N][K] bf16 sign matrix
    OutT* __restrict__ C,                   // [M][N]
    const float* __restrict__ beta_ptr,
    int N, int K) {
  __shared__ __align__(16) unsigned short ldsA[2][BM * LDSROW];  // 2 x 10 KB
  __shared__ __align__(16) unsigned short ldsB[2][BN * LDSROW];  // 2 x 20 KB

  const int tid  = threadIdx.x;
  const int lane = tid & 31;
  const int wave = tid >> 5;
  const int wm   = wave >> 2;  // 0..1  -> 64-row slice
  const int wn   = wave & 3;   // 0..3  -> 64-col slice
  const int m0   = blockIdx.y * BM;
  const int n0   = blockIdx.x * BN;

  v8f acc[4][4] = {};

  auto compute = [&](int buf) {
    const int r16  = lane & 15;
    const int asel = lane >> 4;  // 0 or 1
    v16bf afr[4], bfr[4];
#pragma unroll
    for (int mi = 0; mi < 4; ++mi) {
      // A 16x32 bf16 frag: lane m=r16; K groups {8*asel.., 16+8*asel..}
      const unsigned short* p =
          &ldsA[buf][(wm * 64 + mi * 16 + r16) * LDSROW + asel * 8];
      v4u lo = *(const v4u*)p;
      v4u hi = *(const v4u*)(p + 16);
      afr[mi] = frag_cast(lo, hi);
    }
#pragma unroll
    for (int ni = 0; ni < 4; ++ni) {
      // B 32x16 bf16 frag: lane n=r16; K groups {8*asel.., 16+8*asel..}
      const unsigned short* p =
          &ldsB[buf][(wn * 64 + ni * 16 + r16) * LDSROW + asel * 8];
      v4u lo = *(const v4u*)p;
      v4u hi = *(const v4u*)(p + 16);
      bfr[ni] = frag_cast(lo, hi);
    }
#pragma unroll
    for (int mi = 0; mi < 4; ++mi)
#pragma unroll
      for (int ni = 0; ni < 4; ++ni)
        acc[mi][ni] = __builtin_amdgcn_wmma_f32_16x16x32_bf16(
            false, afr[mi], false, bfr[ni], (short)0, acc[mi][ni], false, false);
  };

  const int nK = K / BK;

#if USE_TDM
  // -------- Tensor Data Mover staging: wave 0 drives the DMA ---------------
  auto issue = [&](int buf, int kt) {
    tdm_load_tile(A + (size_t)m0 * K + (size_t)kt * BK,
                  lds_offset(&ldsA[buf][0]), BM, K);
    tdm_load_tile(Bm + (size_t)n0 * K + (size_t)kt * BK,
                  lds_offset(&ldsB[buf][0]), BN, K);
  };
  if (wave == 0) issue(0, 0);
  for (int kt = 0; kt < nK; ++kt) {
    if (wave == 0) {
      if (kt + 1 < nK) {
        issue((kt + 1) & 1, kt + 1);
        // 4 tensor ops outstanding; in-order completion per wave, so <=2
        // guarantees the current buffer's pair has landed in LDS.
        __builtin_amdgcn_s_wait_tensorcnt(2);
      } else {
        __builtin_amdgcn_s_wait_tensorcnt(0);
      }
    }
    __syncthreads();          // publish buffer kt to all waves
    compute(kt & 1);
    __syncthreads();          // computing done before buffer kt is re-filled
  }
#else
  // -------- fallback: global_load_b128 -> ds_store_b128 staging ------------
  const int ldr = tid >> 1;            // A: 2 threads per row, 32B each
  const int ldc = (tid & 1) * 16;
  const size_t aBase = (size_t)(m0 + ldr) * K + ldc;
  const size_t bBase = (size_t)(n0 + tid) * K;  // B: 1 thread per row, 64B
  auto stage = [&](int buf, int kt) {
    const v4u* ga = (const v4u*)(A + aBase + (size_t)kt * BK);
    v4u a0 = ga[0], a1 = ga[1];
    const v4u* gb = (const v4u*)(Bm + bBase + (size_t)kt * BK);
    v4u b0 = gb[0], b1 = gb[1], b2 = gb[2], b3 = gb[3];
    v4u* la = (v4u*)&ldsA[buf][ldr * LDSROW + ldc];
    la[0] = a0; la[1] = a1;
    v4u* lb = (v4u*)&ldsB[buf][tid * LDSROW];
    lb[0] = b0; lb[1] = b1; lb[2] = b2; lb[3] = b3;
  };
  stage(0, 0);
  __syncthreads();
  for (int kt = 0; kt < nK; ++kt) {
    if (kt + 2 < nK) {  // -> global_prefetch_b8
      __builtin_prefetch(A + aBase + (size_t)(kt + 2) * BK, 0, 1);
      __builtin_prefetch(Bm + bBase + (size_t)(kt + 2) * BK, 0, 1);
    }
    if (kt + 1 < nK) stage((kt + 1) & 1, kt + 1);
    compute(kt & 1);
    __syncthreads();
  }
#endif

  const float beta = beta_ptr[0];
#pragma unroll
  for (int mi = 0; mi < 4; ++mi) {
#pragma unroll
    for (int ni = 0; ni < 4; ++ni) {
#pragma unroll
      for (int v = 0; v < 8; ++v) {
        const int m = m0 + wm * 64 + mi * 16 + ((lane >> 4) << 3) + v;
        const int n = n0 + wn * 64 + ni * 16 + (lane & 15);
        float val = acc[mi][ni][v] * beta;
        if (GELU) val = 0.5f * val * (1.0f + erff(val * 0.70710678118654752f));
        if constexpr (sizeof(OutT) == 2) {
          ((unsigned short*)C)[(size_t)m * N + n] = f2bf(val);
        } else {
          ((float*)C)[(size_t)m * N + n] = val;
        }
      }
    }
  }
}

// ---------------------------------------------------------------------------
// Host launch
// ---------------------------------------------------------------------------
extern "C" void kernel_launch(void* const* d_in, const int* in_sizes, int n_in,
                              void* d_out, int out_size, void* d_ws, size_t ws_size,
                              hipStream_t stream) {
  (void)in_sizes; (void)n_in; (void)out_size; (void)ws_size;
  const float* x  = (const float*)d_in[0];  // [8192][2048]
  const float* w1 = (const float*)d_in[1];  // [8192][2048]
  const float* w2 = (const float*)d_in[2];  // [2048][8192]
  float* out = (float*)d_out;               // [8192][2048]

  char* ws = (char*)d_ws;
  float* scalars  = (float*)ws;              // [0]=mean1 [1]=beta1 [2]=mean2 [3]=beta2
  float* partials = (float*)(ws + 256);      // 1024 * 2 floats
  unsigned short* xhat = (unsigned short*)(ws + 16384);
  unsigned short* w1s  = xhat + (size_t)M_DIM * D_DIM;           // 32 MiB each
  unsigned short* w2s  = w1s  + (size_t)F_DIM * D_DIM;
  unsigned short* h    = w2s  + (size_t)D_DIM * F_DIM;           // 128 MiB

  const int nW = F_DIM * D_DIM;  // 16,777,216 (same for w1 and w2)

  // weight stats + sign quantization
  reduce_w<<<1024, 256, 0, stream>>>(w1, nW, partials);
  finalize_w<<<1, 256, 0, stream>>>(partials, 1024, 1.0f / (float)nW, scalars + 0);
  quant_w<<<nW / 1024, 256, 0, stream>>>(w1, scalars + 0, w1s, nW);
  reduce_w<<<1024, 256, 0, stream>>>(w2, nW, partials);
  finalize_w<<<1, 256, 0, stream>>>(partials, 1024, 1.0f / (float)nW, scalars + 2);
  quant_w<<<nW / 1024, 256, 0, stream>>>(w2, scalars + 2, w2s, nW);

  // activation fake-quant of x
  act_quant_x<<<M_DIM, 256, 0, stream>>>(x, xhat);

  // h = gelu(beta1 * xhat @ w1s^T)   [8192 x 8192] bf16
  gemm_bitlinear<true, unsigned short>
      <<<dim3(F_DIM / BN, M_DIM / BM), 256, 0, stream>>>(
          xhat, w1s, h, scalars + 1, F_DIM, D_DIM);

  // layernorm + absmax of h, in place
  act_quant_h<<<M_DIM, 256, 0, stream>>>(h);

  // out = beta2 * h_hat @ w2s^T      [8192 x 2048] fp32
  gemm_bitlinear<false, float>
      <<<dim3(D_DIM / BN, M_DIM / BM), 256, 0, stream>>>(
          h, w2s, out, scalars + 3, D_DIM, F_DIM);
}
